// Attention_33354716021131
// MI455X (gfx1250) — compile-verified, matrix operand order checked by reading, not run on hardware
//
#include <hip/hip_runtime.h>
#include <hip/hip_bf16.h>

// ---------------- problem constants ----------------
constexpr int DIMSZ   = 4096;
constexpr int NHEADS  = 32;
constexpr int NKV     = 8;
constexpr int HD      = 128;
constexpr int BATCH   = 2;
constexpr int SEQ     = 2048;
constexpr int ROWS    = BATCH * SEQ;          // 4096

typedef __attribute__((ext_vector_type(16))) __bf16 v16bf;
typedef __attribute__((ext_vector_type(8)))  __bf16 v8bf;
typedef __attribute__((ext_vector_type(8)))  float  v8f;

// -------- gfx1250 async global->LDS copy (guarded; falls back to sync) -----
#ifdef __has_builtin
#if __has_builtin(__builtin_amdgcn_global_load_async_to_lds_b128)
#define HAVE_ASYNC_LDS 1
#endif
#endif
#ifndef HAVE_ASYNC_LDS
#define HAVE_ASYNC_LDS 0
#endif

#if HAVE_ASYNC_LDS
typedef int v4i_vec __attribute__((vector_size(16)));
typedef __attribute__((address_space(1))) v4i_vec* as1_v4i;
typedef __attribute__((address_space(3))) v4i_vec* as3_v4i;
__device__ inline void async_copy32(const __bf16* g, __bf16* l) {
    __builtin_amdgcn_global_load_async_to_lds_b128((as1_v4i)(void*)g,       (as3_v4i)(void*)l,       0, 0);
    __builtin_amdgcn_global_load_async_to_lds_b128((as1_v4i)(void*)(g + 8), (as3_v4i)(void*)(l + 8), 0, 0);
}
__device__ inline void wait_async_2() {
#if __has_builtin(__builtin_amdgcn_s_wait_asynccnt)
    __builtin_amdgcn_s_wait_asynccnt(2);
#else
    asm volatile("s_wait_asynccnt 0x2" ::: "memory");
#endif
}
__device__ inline void wait_async_0() {
#if __has_builtin(__builtin_amdgcn_s_wait_asynccnt)
    __builtin_amdgcn_s_wait_asynccnt(0);
#else
    asm volatile("s_wait_asynccnt 0x0" ::: "memory");
#endif
}
#endif

// Load one 16x32 bf16 WMMA operand fragment (A layout, or B layout from
// K-contiguous [N,K]-major data). Per CDNA5 ISA 16-bit operand layout:
//   lane 0-15 : row = lane,      K = {0..7, 16..23}
//   lane16-31 : row = lane-16,   K = {8..15, 24..31}
__device__ inline v16bf load_frag(const __bf16* __restrict__ base, int stride, int lane) {
    int r    = lane & 15;
    int half = lane >> 4;
    const __bf16* p = base + (size_t)r * stride + half * 8;
    v8bf lo = *reinterpret_cast<const v8bf*>(p);
    v8bf hi = *reinterpret_cast<const v8bf*>(p + 16);
    v16bf out;
#pragma unroll
    for (int i = 0; i < 8; ++i) { out[i] = lo[i]; out[i + 8] = hi[i]; }
    return out;
}

// ---------------- conversion kernels ----------------
__global__ void cvt_f32_bf16(const float* __restrict__ src, __bf16* __restrict__ dst, int n) {
    int i = blockIdx.x * blockDim.x + threadIdx.x;
    if (i < n) dst[i] = (__bf16)src[i];
}

// W [K,N] f32 (row-major) -> Wt [N,K] bf16 (row-major)
__global__ void cvt_transpose_bf16(const float* __restrict__ src, __bf16* __restrict__ dst,
                                   int K, int N) {
    int i = blockIdx.x * blockDim.x + threadIdx.x;
    if (i >= K * N) return;
    int k = i / N;
    int n = i % N;
    dst[(size_t)n * K + k] = (__bf16)src[i];
}

// RoPE + convert + head-major transpose.
// src: [B*S, nh*128] f32   dst: [B, nh, S, 128] bf16
__global__ void rope_cvt(const float* __restrict__ src, __bf16* __restrict__ dst, int nh) {
    int idx = blockIdx.x * blockDim.x + threadIdx.x;
    int total = BATCH * SEQ * nh * 64;
    if (idx >= total) return;
    int j = idx & 63;
    int t = idx >> 6;
    int h = t % nh;  t /= nh;
    int s = t % SEQ;
    int b = t / SEQ;
    const float lncoef = 0.14391156831212787f; // ln(10000)/64
    float ang = (float)s * __expf(-(float)j * lncoef);
    float sn, cs;
    __sincosf(ang, &sn, &cs);
    size_t soff = (((size_t)(b * SEQ + s) * nh) + h) * HD + 2 * j;
    float x0 = src[soff], x1 = src[soff + 1];
    size_t doff = (((size_t)(b * nh + h) * SEQ) + s) * HD + 2 * j;
    dst[doff]     = (__bf16)(x0 * cs - x1 * sn);
    dst[doff + 1] = (__bf16)(x0 * sn + x1 * cs);
}

// v: [B*S, KV*128] f32 -> [B, KV, 128, S] bf16
__global__ void v_cvt_T(const float* __restrict__ src, __bf16* __restrict__ dst) {
    int idx = blockIdx.x * blockDim.x + threadIdx.x;
    int total = BATCH * SEQ * NKV * HD;
    if (idx >= total) return;
    int d  = idx & (HD - 1);
    int t  = idx >> 7;
    int kv = t % NKV; t /= NKV;
    int s  = t % SEQ;
    int b  = t / SEQ;
    dst[(((size_t)(b * NKV + kv) * HD) + d) * SEQ + s] = (__bf16)src[idx];
}

// ---------------- bf16 WMMA GEMM ----------------
// C[M,N] f32 = A[M,K] bf16 (row-major) @ Bt[N,K] bf16 (B transposed).
// Block tile 128x128, 8 waves, wave tile 16x128, K-step 32.
// B panel double-buffered in LDS via async global->LDS copies when available.
__global__ __launch_bounds__(256) void gemm_bf16_wmma(
    const __bf16* __restrict__ A, const __bf16* __restrict__ Bt,
    float* __restrict__ C, int M, int N, int K) {
    __shared__ __bf16 ldsB[2][128 * 32];   // [buf][n 0..127][k 0..31]

    int tid  = threadIdx.x;
    int wave = tid >> 5, lane = tid & 31;
    int nBlocks = N >> 7;
    int mb = (int)blockIdx.x / nBlocks;
    int nb = (int)blockIdx.x % nBlocks;
    int m0 = mb << 7, n0 = nb << 7;
    int mw = m0 + wave * 16;

    v8f acc[8];
#pragma unroll
    for (int t = 0; t < 8; ++t)
#pragma unroll
        for (int i = 0; i < 8; ++i) acc[t][i] = 0.0f;

    // per-thread staging assignment: 16 bf16 (= 2 x b128) of the Bt panel
    int nrow = tid >> 1;
    int kk   = (tid & 1) * 16;
    const __bf16* bsrc = Bt + (size_t)(n0 + nrow) * K + kk;
    const __bf16* Arow = A + (size_t)mw * K;
    int ntiles = K >> 5;

#if HAVE_ASYNC_LDS
    async_copy32(bsrc, &ldsB[0][nrow * 32 + kk]);          // prologue: tile 0
#endif

    for (int i = 0; i < ntiles; ++i) {
        int k0 = i << 5;
#if HAVE_ASYNC_LDS
        int cur = i & 1;
        if (i + 1 < ntiles) {
            async_copy32(bsrc + (size_t)(i + 1) * 32, &ldsB[cur ^ 1][nrow * 32 + kk]);
            wait_async_2();                                 // tile i complete (in-order)
        } else {
            wait_async_0();
        }
        __syncthreads();                                    // whole panel visible
        const __bf16* bpanel = &ldsB[cur][0];
#else
        {
            const v8bf* src = reinterpret_cast<const v8bf*>(bsrc + k0);
            v8bf* dst = reinterpret_cast<v8bf*>(&ldsB[0][nrow * 32 + kk]);
            dst[0] = src[0];
            dst[1] = src[1];
        }
        __syncthreads();
        const __bf16* bpanel = &ldsB[0][0];
#endif
        v16bf af = load_frag(Arow + k0, K, lane);
        v16bf bfr[8];
#pragma unroll
        for (int nt = 0; nt < 8; ++nt)
            bfr[nt] = load_frag(bpanel + (nt * 16) * 32, 32, lane);
#pragma unroll
        for (int nt = 0; nt < 8; ++nt)
            acc[nt] = __builtin_amdgcn_wmma_f32_16x16x32_bf16(
                false, af, false, bfr[nt], (short)0, acc[nt], false, false);
        __syncthreads();                                    // before panel reuse
    }

    int r = lane & 15, half = lane >> 4;
#pragma unroll
    for (int nt = 0; nt < 8; ++nt)
#pragma unroll
        for (int j = 0; j < 8; ++j) {
            int row = mw + j + half * 8;
            int col = n0 + nt * 16 + r;
            C[(size_t)row * N + col] = acc[nt][j];
        }
}

// ---------------- flash attention (causal, GQA) ----------------
// Q [B,H,S,128] bf16, K [B,KV,S,128] bf16, Vt [B,KV,128,S] bf16,
// O [B,S,H*128] bf16. Block = 8 waves; wave owns 16 query rows.
__global__ __launch_bounds__(256) void flash_attn_wmma(
    const __bf16* __restrict__ Q, const __bf16* __restrict__ Kc,
    const __bf16* __restrict__ Vt, __bf16* __restrict__ O) {
    __shared__ __bf16 ldsP[8][16][40];   // per-wave P scratch, padded rows (80B)

    int tid  = threadIdx.x;
    int wave = tid >> 5, lane = tid & 31;
    int r = lane & 15, half = lane >> 4;

    int bid = blockIdx.x;
    int qt  = bid & (SEQ / 128 - 1);
    int h   = (bid / (SEQ / 128)) % NHEADS;
    int b   = bid / ((SEQ / 128) * NHEADS);
    int kv  = h / (NHEADS / NKV);
    int q0  = qt * 128 + wave * 16;

    const __bf16* qbase = Q  + (((size_t)(b * NHEADS + h) * SEQ) + q0) * HD;
    const __bf16* kbase = Kc + ((size_t)(b * NKV + kv)) * SEQ * HD;
    const __bf16* vbase = Vt + ((size_t)(b * NKV + kv)) * HD * SEQ;

    v16bf qf[4];
#pragma unroll
    for (int d = 0; d < 4; ++d) qf[d] = load_frag(qbase + d * 32, HD, lane);

    v8f acc[8];
    float mrow[8], lrow[8];
#pragma unroll
    for (int j = 0; j < 8; ++j) {
        mrow[j] = -1e30f;
        lrow[j] = 0.0f;
#pragma unroll
        for (int i = 0; i < 8; ++i) acc[j][i] = 0.0f;
    }

    const float scale = 0.08838834764831845f;  // 1/sqrt(128)
    int kmax = q0 + 16;                        // keys needed: <= q0+15

    for (int kb = 0; kb < kmax; kb += 32) {
        // prefetch next K tile (gfx1250 global_prefetch)
        __builtin_prefetch(kbase + (size_t)(kb + 32) * HD + lane * 32, 0, 0);
#pragma unroll
        for (int sub = 0; sub < 2; ++sub) {
            int key0 = kb + sub * 16;
            v8f s = {0.f, 0.f, 0.f, 0.f, 0.f, 0.f, 0.f, 0.f};
            v16bf kfr[4];
#pragma unroll
            for (int d = 0; d < 4; ++d)
                kfr[d] = load_frag(kbase + (size_t)key0 * HD + d * 32, HD, lane);
#pragma unroll
            for (int d = 0; d < 4; ++d)
                s = __builtin_amdgcn_wmma_f32_16x16x32_bf16(
                    false, qf[d], false, kfr[d], (short)0, s, false, false);
            // online softmax per row (row lives across 16 lanes of one VGPR)
#pragma unroll
            for (int j = 0; j < 8; ++j) {
                int qrow = q0 + j + half * 8;
                int key  = key0 + r;
                float sv = s[j] * scale;
                if (key > qrow) sv = -1e30f;
                float mx = sv;
                mx = fmaxf(mx, __shfl_xor(mx, 1));
                mx = fmaxf(mx, __shfl_xor(mx, 2));
                mx = fmaxf(mx, __shfl_xor(mx, 4));
                mx = fmaxf(mx, __shfl_xor(mx, 8));
                float mN    = fmaxf(mrow[j], mx);
                float alpha = __expf(mrow[j] - mN);
                mrow[j] = mN;
                float p  = __expf(sv - mN);
                float ps = p;
                ps += __shfl_xor(ps, 1);
                ps += __shfl_xor(ps, 2);
                ps += __shfl_xor(ps, 4);
                ps += __shfl_xor(ps, 8);
                lrow[j] = lrow[j] * alpha + ps;
#pragma unroll
                for (int nt = 0; nt < 8; ++nt) acc[nt][j] *= alpha;
                ldsP[wave][j + half * 8][sub * 16 + r] = (__bf16)p;
            }
        }
        // P(16x32) @ V(32x128): A-frag from LDS, B-frags from pre-transposed Vt
        v16bf pf = load_frag(&ldsP[wave][0][0], 40, lane);
#pragma unroll
        for (int g = 0; g < 2; ++g) {
            v16bf vfr[4];
#pragma unroll
            for (int t = 0; t < 4; ++t)
                vfr[t] = load_frag(vbase + (size_t)((g * 4 + t) * 16) * SEQ + kb, SEQ, lane);
#pragma unroll
            for (int t = 0; t < 4; ++t)
                acc[g * 4 + t] = __builtin_amdgcn_wmma_f32_16x16x32_bf16(
                    false, pf, false, vfr[t], (short)0, acc[g * 4 + t], false, false);
        }
    }

    // epilogue: normalize and store to [B,S,H*128] bf16 (GEMM-ready layout)
#pragma unroll
    for (int j = 0; j < 8; ++j) {
        float inv = 1.0f / lrow[j];
        int srow  = q0 + j + half * 8;
        __bf16* orow = O + ((size_t)b * SEQ + srow) * (NHEADS * HD) + h * HD;
#pragma unroll
        for (int nt = 0; nt < 8; ++nt)
            orow[nt * 16 + r] = (__bf16)(acc[nt][j] * inv);
    }
}

// ---------------- launcher ----------------
extern "C" void kernel_launch(void* const* d_in, const int* in_sizes, int n_in,
                              void* d_out, int out_size, void* d_ws, size_t ws_size,
                              hipStream_t stream) {
    const float* x  = (const float*)d_in[0];
    const float* wq = (const float*)d_in[1];
    const float* wk = (const float*)d_in[2];
    const float* wv = (const float*)d_in[3];
    const float* wo = (const float*)d_in[4];
    float* out = (float*)d_out;

    char* ws = (char*)d_ws;
    const size_t MB = 1ull << 20;
    __bf16* xb   = (__bf16*)(ws + 0);          // 32 MB  x bf16 [4096,4096]
    __bf16* wqT  = (__bf16*)(ws + 32 * MB);    // 32 MB  wq^T  [4096,4096]
    __bf16* wkT  = (__bf16*)(ws + 64 * MB);    //  8 MB  wk^T  [1024,4096]
    __bf16* wvT  = (__bf16*)(ws + 72 * MB);    //  8 MB  wv^T  [1024,4096]
    __bf16* woT  = (__bf16*)(ws + 80 * MB);    // 32 MB  wo^T  [4096,4096]
    float*  qf   = (float*) (ws + 112 * MB);   // 64 MB  q f32 [4096,4096]
    float*  kf   = (float*) (ws + 176 * MB);   // 16 MB  k f32 [4096,1024]
    float*  vf   = (float*) (ws + 192 * MB);   // 16 MB  v f32 [4096,1024]
    __bf16* qb   = (__bf16*)(ws + 208 * MB);   // 32 MB  q bf16 [B,H,S,128]
    __bf16* kbh  = (__bf16*)(ws + 240 * MB);   //  8 MB  k bf16 [B,KV,S,128]
    __bf16* vtb  = (__bf16*)(ws + 248 * MB);   //  8 MB  v bf16 [B,KV,128,S]
    __bf16* ab   = (__bf16*)(ws + 256 * MB);   // 32 MB  attn out bf16 [4096,4096]

    const int TPB = 256;
    {
        int n = ROWS * DIMSZ;
        cvt_f32_bf16<<<(n + TPB - 1) / TPB, TPB, 0, stream>>>(x, xb, n);
    }
    cvt_transpose_bf16<<<(DIMSZ * DIMSZ + TPB - 1) / TPB, TPB, 0, stream>>>(wq, wqT, DIMSZ, DIMSZ);
    cvt_transpose_bf16<<<(DIMSZ * 1024 + TPB - 1) / TPB, TPB, 0, stream>>>(wk, wkT, DIMSZ, 1024);
    cvt_transpose_bf16<<<(DIMSZ * 1024 + TPB - 1) / TPB, TPB, 0, stream>>>(wv, wvT, DIMSZ, 1024);
    cvt_transpose_bf16<<<(DIMSZ * DIMSZ + TPB - 1) / TPB, TPB, 0, stream>>>(wo, woT, DIMSZ, DIMSZ);

    gemm_bf16_wmma<<<(ROWS / 128) * (DIMSZ / 128), TPB, 0, stream>>>(xb, wqT, qf, ROWS, DIMSZ, DIMSZ);
    gemm_bf16_wmma<<<(ROWS / 128) * (1024 / 128), TPB, 0, stream>>>(xb, wkT, kf, ROWS, 1024, DIMSZ);
    gemm_bf16_wmma<<<(ROWS / 128) * (1024 / 128), TPB, 0, stream>>>(xb, wvT, vf, ROWS, 1024, DIMSZ);

    {
        int nq = BATCH * SEQ * NHEADS * 64;
        rope_cvt<<<(nq + TPB - 1) / TPB, TPB, 0, stream>>>(qf, qb, NHEADS);
        int nk = BATCH * SEQ * NKV * 64;
        rope_cvt<<<(nk + TPB - 1) / TPB, TPB, 0, stream>>>(kf, kbh, NKV);
        int nv = BATCH * SEQ * NKV * HD;
        v_cvt_T<<<(nv + TPB - 1) / TPB, TPB, 0, stream>>>(vf, vtb);
    }

    flash_attn_wmma<<<BATCH * NHEADS * (SEQ / 128), TPB, 0, stream>>>(qb, kbh, vtb, ab);

    gemm_bf16_wmma<<<(ROWS / 128) * (DIMSZ / 128), TPB, 0, stream>>>(ab, woT, out, ROWS, DIMSZ, DIMSZ);
}
